// CrossAttention_66133906424271
// MI455X (gfx1250) — matrix operand order repaired and checked by reading, not compiled, not run on hardware
//
#include <hip/hip_runtime.h>

// ---------------------------------------------------------------------------
// CDNA5 (gfx1250) cross-attention, bf16 WMMA pipeline with async global->LDS
// double-buffered GEMMs.
// Fragment layouts per CDNA5 ISA (wave32):
//   A (16x32 bf16): lane l, row M=l%16; K-chunk  [sel*16 .. +16) contiguous (this code's order, same as B), sel=l/16
//   B (32x16 bf16): lane l, col N=l%16; K-chunk  [sel*16 .. +16) contiguous (this code's order, same as A)
//   C/D (16x16 f32): lane l, col N=l%16; rows M = sel*8 + vgpr(0..7)
// ---------------------------------------------------------------------------

typedef unsigned short u16;
typedef __attribute__((ext_vector_type(16))) __bf16 v16bf;
typedef __attribute__((ext_vector_type(8)))  float  v8f;
typedef __attribute__((ext_vector_type(4)))  int    v4i;

// address-space qualified int4 for the async-to-LDS builtin
typedef __attribute__((address_space(1))) v4i v4i_g;   // global
typedef __attribute__((address_space(3))) v4i v4i_l;   // LDS

#if defined(__gfx1250__) && __has_builtin(__builtin_amdgcn_global_load_async_to_lds_b128)
#define USE_ASYNC 1
#else
#define USE_ASYNC 0
#endif

__device__ __forceinline__ u16 f2bf(float f) {
    unsigned u = __builtin_bit_cast(unsigned, f);
    u += 0x7FFFu + ((u >> 16) & 1u);            // round-to-nearest-even
    return (u16)(u >> 16);
}

__device__ __forceinline__ v16bf mkfrag(uint4 lo, uint4 hi) {
    union { uint4 u[2]; v16bf v; } t;
    t.u[0] = lo; t.u[1] = hi;
    return t.v;
}

__device__ __forceinline__ v8f wmma_bf16(v16bf a, v16bf b, v8f c) {
    return __builtin_amdgcn_wmma_f32_16x16x32_bf16(
        /*neg_a=*/false, a, /*neg_b=*/false, b,
        /*c_mod=*/(short)0, c, /*reuse_a=*/false, /*reuse_b=*/false);
}

// async 16-byte global -> LDS copy (GLOBAL_LOAD_ASYNC_TO_LDS_B128, ASYNCcnt)
__device__ __forceinline__ void async_b128(const void* g, void* l) {
#if USE_ASYNC
    // generic->AS pointer via integer round-trip (LDS offset lives in addr[31:0])
    v4i_g* gp = (v4i_g*)(unsigned long long)g;
    v4i_l* lp = (v4i_l*)(unsigned)(unsigned long long)l;
    __builtin_amdgcn_global_load_async_to_lds_b128(gp, lp, 0, 0);
#else
    *(uint4*)l = *(const uint4*)g;
#endif
}

__device__ __forceinline__ void wait_async() {
#if USE_ASYNC
#if __has_builtin(__builtin_amdgcn_s_wait_asynccnt)
    __builtin_amdgcn_s_wait_asynccnt(0);
#else
    asm volatile("s_wait_asynccnt 0" ::: "memory");
#endif
#endif
}

// ---------------------------------------------------------------------------
// fp32 -> bf16 streaming convert (8 elems / thread)
// ---------------------------------------------------------------------------
__global__ __launch_bounds__(256) void convert_bf16(const float* __restrict__ in,
                                                    u16* __restrict__ out, int n8) {
    int i = blockIdx.x * 256 + threadIdx.x;
    if (i < n8) {
        const float4* p = (const float4*)in + (size_t)i * 2;
        float4 f0 = p[0], f1 = p[1];
        uint4 d;
        d.x = (unsigned)f2bf(f0.x) | ((unsigned)f2bf(f0.y) << 16);
        d.y = (unsigned)f2bf(f0.z) | ((unsigned)f2bf(f0.w) << 16);
        d.z = (unsigned)f2bf(f1.x) | ((unsigned)f2bf(f1.y) << 16);
        d.w = (unsigned)f2bf(f1.z) | ((unsigned)f2bf(f1.w) << 16);
        ((uint4*)out)[i] = d;
    }
}

// ---------------------------------------------------------------------------
// Weight transpose + fp32->bf16: W[K][N] -> WT[N][K]
// ---------------------------------------------------------------------------
__global__ __launch_bounds__(256) void transpose_w(const float* __restrict__ W,
                                                   u16* __restrict__ WT,
                                                   int K, int N, int total) {
    int i = blockIdx.x * 256 + threadIdx.x;
    if (i < total) {
        int n = i / K, k = i % K;
        WT[i] = f2bf(W[(size_t)k * N + n]);
    }
}

// ---------------------------------------------------------------------------
// GEMM: C[M][1024] = A[M][K](bf16) @ BT[1024][K]^T (bf16), f32 accumulate.
// Block: 256 threads (8 waves, 2x4), block tile 128(M) x 256(N), K-step 32.
// Wave tile 64x64 = 4x4 fragments -> 16 WMMAs per K-step per wave.
// Double-buffered LDS, async global->LDS staging overlapped with compute.
// ---------------------------------------------------------------------------
#define BM 128
#define BN 256
#define BK 32

template<bool OUTF32>
__global__ __launch_bounds__(256) void gemm_async(const u16* __restrict__ A,
                                                  const u16* __restrict__ BT,
                                                  void* __restrict__ Cv,
                                                  const float* __restrict__ bias,
                                                  int Mb, int K) {
    __shared__ __align__(16) u16 ldsA[2][BM][BK];   // 2 x 8 KB
    __shared__ __align__(16) u16 ldsB[2][BN][BK];   // 2 x 16 KB

    const int t    = threadIdx.x;
    const int wave = t >> 5, lane = t & 31;
    const int lh   = lane & 15, sel = lane >> 4;
    const int wm   = wave >> 2, wn = wave & 3;      // 2 x 4 wave grid
    const int m0   = blockIdx.x * BM;
    const int n0   = blockIdx.y * BN;

    v8f acc[4][4] = {};

    const int nk = K / BK;

    // stage K-tile kk*BK into LDS buffer bb
    auto stage = [&](int kk, int bb) {
        const int k0 = kk * BK;
        #pragma unroll
        for (int i = 0; i < 2; ++i) {               // A: 128x32 bf16 = 8 KB
            int c = t + i * 256, row = c >> 2, seg = c & 3;
            int gm = m0 + row;
            u16* dst = &ldsA[bb][row][seg * 8];
            if (gm < Mb)
                async_b128(A + (size_t)gm * K + k0 + seg * 8, dst);
            else {
                uint4 z = {0u, 0u, 0u, 0u};
                *(uint4*)dst = z;
            }
        }
        #pragma unroll
        for (int i = 0; i < 4; ++i) {               // B: 256x32 bf16 = 16 KB
            int c = t + i * 256, row = c >> 2, seg = c & 3;
            async_b128(BT + (size_t)(n0 + row) * K + k0 + seg * 8,
                       &ldsB[bb][row][seg * 8]);
        }
    };

    stage(0, 0);
    wait_async();
    __syncthreads();

    for (int kt = 0; kt < nk; ++kt) {
        const int cur = kt & 1;
        if (kt + 1 < nk) stage(kt + 1, cur ^ 1);    // overlap next tile's DMA

        v16bf a[4];
        #pragma unroll
        for (int fm = 0; fm < 4; ++fm) {
            const u16* ar = &ldsA[cur][wm * 64 + fm * 16 + lh][0];
            a[fm] = mkfrag(*(const uint4*)(ar + sel * 16),
                           *(const uint4*)(ar + sel * 16 + 8));
        }
        #pragma unroll
        for (int fn = 0; fn < 4; ++fn) {
            const u16* br = &ldsB[cur][wn * 64 + fn * 16 + lh][sel * 16];
            v16bf b = mkfrag(*(const uint4*)br, *(const uint4*)(br + 8));
            #pragma unroll
            for (int fm = 0; fm < 4; ++fm)
                acc[fm][fn] = wmma_bf16(a[fm], b, acc[fm][fn]);
        }

        wait_async();
        __syncthreads();
    }

    // ---- epilogue ----
    #pragma unroll
    for (int fm = 0; fm < 4; ++fm)
        #pragma unroll
        for (int fn = 0; fn < 4; ++fn)
            #pragma unroll
            for (int r = 0; r < 8; ++r) {
                int m = m0 + wm * 64 + fm * 16 + sel * 8 + r;
                int n = n0 + wn * 64 + fn * 16 + lh;
                if (m < Mb) {
                    if (OUTF32)
                        ((float*)Cv)[(size_t)m * 1024 + n] = acc[fm][fn][r] + bias[n];
                    else
                        ((u16*)Cv)[(size_t)m * 1024 + n] = f2bf(acc[fm][fn][r]);
                }
            }
}

// ---------------------------------------------------------------------------
// v[620][1024] (row-major per (b,m), cols h*64+d) -> vT[128 bh][64 d][96 m],
// zero-padded for m >= 77 (so padded K never pollutes P@V).
// ---------------------------------------------------------------------------
__global__ __launch_bounds__(256) void build_vT(const u16* __restrict__ v,
                                                u16* __restrict__ vT) {
    int i = blockIdx.x * 256 + threadIdx.x;
    if (i < 128 * 64 * 96) {
        int bh = i / (64 * 96), rem = i % (64 * 96);
        int d = rem / 96, m = rem % 96;
        int b = bh >> 4, h = bh & 15;
        vT[i] = (m < 77) ? v[(size_t)(b * 77 + m) * 1024 + h * 64 + d] : (u16)0;
    }
}

// ---------------------------------------------------------------------------
// Attention: per wave, 16 q-rows of one (b,h). S = q k^T (K=64, j padded to
// 80), softmax over j<77, P (bf16, zero-padded K=96) @ vT -> 16x64 output.
// Block: 128 threads = 4 waves, each an independent 16-row tile.
// ---------------------------------------------------------------------------
__global__ __launch_bounds__(128) void attn_kernel(const u16* __restrict__ q,
                                                   const u16* __restrict__ kbuf,
                                                   const u16* __restrict__ vT,
                                                   u16* __restrict__ o) {
    __shared__ __align__(16) float S[4][16 * 80];
    __shared__ __align__(16) u16   P[4][16 * 96];

    const int t    = threadIdx.x;
    const int wave = t >> 5, lane = t & 31;
    const int lh   = lane & 15, sel = lane >> 4;
    const int bh   = blockIdx.x, b = bh >> 4, h = bh & 15;
    const int n0   = blockIdx.y * 64 + wave * 16;

    float* Sw = S[wave];
    u16*   Pw = P[wave];

    // q A-fragments (K = d: two 32-wide steps)
    const u16* qp = q + ((size_t)(b * 4096 + n0 + lh)) * 1024 + h * 64;
    v16bf aq0 = mkfrag(*(const uint4*)(qp + sel * 16),
                       *(const uint4*)(qp + sel * 16 + 8));
    v16bf aq1 = mkfrag(*(const uint4*)(qp + 32 + sel * 16),
                       *(const uint4*)(qp + 32 + sel * 16 + 8));

    // scores: 5 j-tiles of 16 (77 -> 80, masked later)
    #pragma unroll
    for (int f = 0; f < 5; ++f) {
        int j = f * 16 + lh;
        const u16* kr = kbuf + ((size_t)(b * 77 + j)) * 1024 + h * 64 + sel * 16;
        v16bf bk0 = mkfrag(*(const uint4*)kr, *(const uint4*)(kr + 8));
        v16bf bk1 = mkfrag(*(const uint4*)(kr + 32), *(const uint4*)(kr + 40));
        v8f acc = {};
        acc = wmma_bf16(aq0, bk0, acc);
        acc = wmma_bf16(aq1, bk1, acc);
        #pragma unroll
        for (int r = 0; r < 8; ++r)
            Sw[(sel * 8 + r) * 80 + f * 16 + lh] = acc[r];
    }
    __syncthreads();

    // softmax: lanes 0..15 each own one row; j >= 77 ignored, padded P = 0
    if (lane < 16) {
        float* row = Sw + lane * 80;
        float mx = -3.4e38f;
        for (int j = 0; j < 77; ++j) {
            float s = row[j] * 0.125f;          // scale = D^-0.5 = 1/8
            row[j] = s;
            mx = fmaxf(mx, s);
        }
        float sum = 0.f;
        for (int j = 0; j < 77; ++j) {
            float e = __expf(row[j] - mx);
            row[j] = e;
            sum += e;
        }
        float inv = 1.f / sum;
        u16* pr = Pw + lane * 96;
        for (int j = 0; j < 77; ++j) pr[j] = f2bf(row[j] * inv);
        for (int j = 77; j < 96; ++j) pr[j] = 0;
    }
    __syncthreads();

    // P @ V: K = 96 (3 steps), N = 64 (4 tiles)
    v8f O[4] = {};
    #pragma unroll
    for (int ks = 0; ks < 3; ++ks) {
        const u16* pr = Pw + lh * 96 + ks * 32 + sel * 16;
        v16bf ap = mkfrag(*(const uint4*)pr, *(const uint4*)(pr + 8));
        #pragma unroll
        for (int f = 0; f < 4; ++f) {
            const u16* vr = vT + ((size_t)bh * 64 + f * 16 + lh) * 96 + ks * 32 + sel * 16;
            v16bf bv = mkfrag(*(const uint4*)vr, *(const uint4*)(vr + 8));
            O[f] = wmma_bf16(ap, bv, O[f]);
        }
    }

    #pragma unroll
    for (int f = 0; f < 4; ++f)
        #pragma unroll
        for (int r = 0; r < 8; ++r) {
            size_t row = (size_t)(b * 4096 + n0 + sel * 8 + r);
            o[row * 1024 + h * 64 + f * 16 + lh] = f2bf(O[f][r]);
        }
}

// ---------------------------------------------------------------------------
// Host launcher. Inputs: x, context, Wq, Wk, Wv, Wo, bo (all fp32).
// ---------------------------------------------------------------------------
extern "C" void kernel_launch(void* const* d_in, const int* in_sizes, int n_in,
                              void* d_out, int out_size, void* d_ws, size_t ws_size,
                              hipStream_t stream) {
    (void)in_sizes; (void)n_in; (void)out_size; (void)ws_size;

    const float* x   = (const float*)d_in[0];   // [8,4096,1024]
    const float* ctx = (const float*)d_in[1];   // [8,77,768]
    const float* Wq  = (const float*)d_in[2];   // [1024,1024]
    const float* Wk  = (const float*)d_in[3];   // [768,1024]
    const float* Wv  = (const float*)d_in[4];   // [768,1024]
    const float* Wo  = (const float*)d_in[5];   // [1024,1024]
    const float* bo  = (const float*)d_in[6];   // [1024]
    float* out = (float*)d_out;                 // [8,4096,1024] fp32

    // workspace layout (bytes, 16B-aligned)
    char* ws = (char*)d_ws;
    size_t oWqT  = 0;
    size_t oWkT  = oWqT  + (size_t)1024 * 1024 * 2;   // 2 MB
    size_t oWvT  = oWkT  + (size_t)1024 * 768 * 2;    // +1.5 MB
    size_t oWoT  = oWvT  + (size_t)1024 * 768 * 2;    // +1.5 MB
    size_t oQ    = oWoT  + (size_t)1024 * 1024 * 2;   // +2 MB
    size_t oK    = oQ    + (size_t)32768 * 1024 * 2;  // +64 MB
    size_t oV    = oK    + (size_t)620 * 1024 * 2;    // +1.21 MB
    size_t oVT   = oV    + (size_t)620 * 1024 * 2;    // +1.21 MB
    size_t oXB   = oVT   + (size_t)128 * 64 * 96 * 2; // +1.5 MB (x bf16; reused as attn out)
    size_t oCtxB = oXB   + (size_t)32768 * 1024 * 2;  // +64 MB
    // total = oCtxB + 0.9 MB  (~140 MB)

    u16* WqT  = (u16*)(ws + oWqT);
    u16* WkT  = (u16*)(ws + oWkT);
    u16* WvT  = (u16*)(ws + oWvT);
    u16* WoT  = (u16*)(ws + oWoT);
    u16* qbuf = (u16*)(ws + oQ);
    u16* kbuf = (u16*)(ws + oK);
    u16* vbuf = (u16*)(ws + oV);
    u16* vT   = (u16*)(ws + oVT);
    u16* xb   = (u16*)(ws + oXB);   // x as bf16; dead after q-proj
    u16* ctxb = (u16*)(ws + oCtxB);
    u16* obuf = xb;                 // attention output aliases x_bf16

    // 1. inputs -> bf16
    convert_bf16<<<(4194304 + 255) / 256, 256, 0, stream>>>(x,   xb,   4194304); // 32768*1024/8
    convert_bf16<<<(59136   + 255) / 256, 256, 0, stream>>>(ctx, ctxb, 59136);   // 616*768/8

    // 2. weights -> bf16, transposed to [N][K]
    transpose_w<<<(1024 * 1024 + 255) / 256, 256, 0, stream>>>(Wq, WqT, 1024, 1024, 1024 * 1024);
    transpose_w<<<(1024 * 768  + 255) / 256, 256, 0, stream>>>(Wk, WkT, 768,  1024, 1024 * 768);
    transpose_w<<<(1024 * 768  + 255) / 256, 256, 0, stream>>>(Wv, WvT, 768,  1024, 1024 * 768);
    transpose_w<<<(1024 * 1024 + 255) / 256, 256, 0, stream>>>(Wo, WoT, 1024, 1024, 1024 * 1024);

    // 3. projections (async double-buffered WMMA GEMMs)
    gemm_async<false><<<dim3(256, 4), 256, 0, stream>>>(xb,   WqT, qbuf, nullptr, 32768, 1024);
    gemm_async<false><<<dim3(5,   4), 256, 0, stream>>>(ctxb, WkT, kbuf, nullptr, 616,   768);
    gemm_async<false><<<dim3(5,   4), 256, 0, stream>>>(ctxb, WvT, vbuf, nullptr, 616,   768);

    // 4. v -> vT [bh][64][96] zero-padded
    build_vT<<<(128 * 64 * 96 + 255) / 256, 256, 0, stream>>>(vbuf, vT);

    // 5. attention (scores + softmax + P@V), all WMMA
    attn_kernel<<<dim3(128, 64), 128, 0, stream>>>(qbuf, kbuf, vT, obuf);

    // 6. output projection with bias, fp32 out
    gemm_async<true><<<dim3(256, 4), 256, 0, stream>>>(obuf, WoT, out, bo, 32768, 1024);
}
